// SoftmaxAttention_22376779612727
// MI455X (gfx1250) — matrix operand order repaired and checked by reading, not compiled, not run on hardware
//
#include <hip/hip_runtime.h>
#include <hip/hip_bf16.h>

// CDNA5 / gfx1250 co-attention (ESIM alignment) kernel.
// out[b,i,:] = softmax_j( Q[b,i,:] . K[b,j,:] ) @ K[b,:,:], run twice with
// Q/K roles swapped. bf16 WMMA (16x16x32) with f32 accumulation, flash-style
// online softmax. K blocks staged in double-buffered LDS (row-major bf16 for
// the S phase, transposed bf16 for the PV phase). Key masking is applied as
// an additive -1e10 penalty (== reference semantics when any key is valid);
// the query-mask term is constant per softmax row and cancels exactly.

typedef __bf16        v16bf __attribute__((ext_vector_type(16)));
typedef float         v8f   __attribute__((ext_vector_type(8)));
typedef unsigned int  v8u   __attribute__((ext_vector_type(8)));

#define NEG_BIG (-1.0e10f)

__device__ __forceinline__ unsigned int pack_bf16x2(float a, float b) {
    unsigned int r;
    asm("v_cvt_pk_bf16_f32 %0, %1, %2" : "=v"(r) : "v"(a), "v"(b));
    return r;
}

// grid: (Nq/32, B), block: 256 threads (8 waves of 32)
__global__ __launch_bounds__(256) void coattn_bf16_wmma(
    const float* __restrict__ Q, const float* __restrict__ K,
    const unsigned char* __restrict__ qmask, const unsigned char* __restrict__ kmask,
    float* __restrict__ Out, int Nq, int Nk, int D)
{
    __shared__ unsigned short Ks [2][32 * 512];  // [buf][key][d]  bf16, 2x32 KB
    __shared__ unsigned short KsT[2][512 * 32];  // [buf][d][key]  bf16, 2x32 KB
    __shared__ float Spart[8 * 256];             // (dh,rt,kt) 16x16 f32 partials, 8 KB
    __shared__ float Kpen[2][32];                // [buf][key] mask penalty (0 / -1e10)

    (void)qmask;  // constant across a softmax row -> cancels in softmax

    const int tid  = threadIdx.x;
    const int w    = tid >> 5;
    const int lane = tid & 31;
    const int h    = lane >> 4;   // lane half
    const int ln   = lane & 15;

    const int b         = blockIdx.y;
    const int row_block = blockIdx.x * 32;

    const int rt  = w & 1;          // row tile (shared by phase A and B)
    const int kt  = (w >> 1) & 1;   // phase A: key sub-tile
    const int dh  = w >> 2;         // phase A: D half
    const int dsl = (w >> 1) * 128; // phase B: D slice base

    const float* Qb = Q + ((size_t)b * Nq) * D;
    const float* Kb = K + ((size_t)b * Nk) * D;
    const unsigned char* kmb = kmask + (size_t)b * Nk;

    // ---- staging: K block (32 keys x 512 d) -> LDS bf16, both layouts ----
    auto stage = [&](int key0, int buf) {
        const int kp = tid & 15;          // key pair (2kp, 2kp+1)
        const int d0 = (tid >> 4) * 32;   // 32 d values per thread
        const float* ka = Kb + (size_t)(key0 + 2 * kp) * D + d0;
        const float* kc = Kb + (size_t)(key0 + 2 * kp + 1) * D + d0;
        unsigned int* dstT = (unsigned int*)&KsT[buf][0];
#pragma unroll
        for (int dd = 0; dd < 32; dd += 4) {
            const float4 va = *(const float4*)(ka + dd);
            const float4 vb = *(const float4*)(kc + dd);
            // transposed copy: KsT[d][key] (key-pair packed per dword)
            dstT[((d0 + dd + 0) * 32 + 2 * kp) >> 1] = pack_bf16x2(va.x, vb.x);
            dstT[((d0 + dd + 1) * 32 + 2 * kp) >> 1] = pack_bf16x2(va.y, vb.y);
            dstT[((d0 + dd + 2) * 32 + 2 * kp) >> 1] = pack_bf16x2(va.z, vb.z);
            dstT[((d0 + dd + 3) * 32 + 2 * kp) >> 1] = pack_bf16x2(va.w, vb.w);
            // row-major copy: Ks[key][d] (d-pair packed per dword)
            uint2 pa2, pb2;
            pa2.x = pack_bf16x2(va.x, va.y); pa2.y = pack_bf16x2(va.z, va.w);
            pb2.x = pack_bf16x2(vb.x, vb.y); pb2.y = pack_bf16x2(vb.z, vb.w);
            *(uint2*)&Ks[buf][(2 * kp)     * 512 + d0 + dd] = pa2;
            *(uint2*)&Ks[buf][(2 * kp + 1) * 512 + d0 + dd] = pb2;
        }
        if (tid < 32) Kpen[buf][tid] = kmb[key0 + tid] ? 0.0f : NEG_BIG;
    };

    // ---- preload this wave's Q A-tiles (16 rows x 32 d per step, bf16) ----
    // A tile, contiguous k order (B uses the same): element e of lane -> M = ln, Kdim = 16h + e
    const int arow = row_block + rt * 16 + ln;
    const float* Qrow = Qb + (size_t)arow * D;
    v16bf qa[8];
#pragma unroll
    for (int s = 0; s < 8; ++s) {
        const int dc = dh * 256 + s * 32;
        const float4 r0 = *(const float4*)(Qrow + dc + 16 * h);
        const float4 r1 = *(const float4*)(Qrow + dc + 16 * h + 4);
        const float4 r2 = *(const float4*)(Qrow + dc + 16 * h + 8);
        const float4 r3 = *(const float4*)(Qrow + dc + 16 * h + 12);
        v8u qw;
        qw[0] = pack_bf16x2(r0.x, r0.y); qw[1] = pack_bf16x2(r0.z, r0.w);
        qw[2] = pack_bf16x2(r1.x, r1.y); qw[3] = pack_bf16x2(r1.z, r1.w);
        qw[4] = pack_bf16x2(r2.x, r2.y); qw[5] = pack_bf16x2(r2.z, r2.w);
        qw[6] = pack_bf16x2(r3.x, r3.y); qw[7] = pack_bf16x2(r3.z, r3.w);
        qa[s] = __builtin_bit_cast(v16bf, qw);
    }

    // ---- output accumulators (16 rows x 128 d, f32 C layout) + softmax stats ----
    v8f o[8];
#pragma unroll
    for (int t = 0; t < 8; ++t) {
#pragma unroll
        for (int v = 0; v < 8; ++v) o[t][v] = 0.0f;
    }
    float mrow = -3.0e38f;  // running row max (this lane owns row ln of its row-tile)
    float lrow = 0.0f;      // running row sum

    const int nkb = Nk >> 5;
    stage(0, 0);  // prologue: stage block 0 into buffer 0

    for (int kb = 0; kb < nkb; ++kb) {
        const int key0 = kb * 32;
        const int cur  = kb & 1;
        __syncthreads();  // staged buffer ready; old buffer's consumers done

        // stage the next block (wrapped on last iteration -> harmless, unread,
        // and race-free since it targets the other buffer). Branchless so the
        // scheduler can interleave these global loads with phase A's WMMAs.
        int knext = key0 + 32;
        if (knext >= Nk) knext = 0;
        stage(knext, cur ^ 1);

        // ---- phase A: partial S tile (16 rows x 16 keys) over this wave's D half ----
        // two independent WMMA accumulation chains (even/odd k-steps) for ILP
        {
            v8f sacc0, sacc1;
#pragma unroll
            for (int v = 0; v < 8; ++v) { sacc0[v] = 0.0f; sacc1[v] = 0.0f; }
            const int keyA = kt * 16 + ln;  // B layout: lane -> N = key (block-local)
            const unsigned short* krow = &Ks[cur][keyA * 512 + dh * 256 + 16 * h];
#pragma unroll
            for (int s2 = 0; s2 < 4; ++s2) {
                const v16bf bt0 = *(const v16bf*)(krow + (2 * s2)     * 32);
                sacc0 = __builtin_amdgcn_wmma_f32_16x16x32_bf16(
                    false, qa[2 * s2],     false, bt0, (short)0, sacc0, false, false);
                const v16bf bt1 = *(const v16bf*)(krow + (2 * s2 + 1) * 32);
                sacc1 = __builtin_amdgcn_wmma_f32_16x16x32_bf16(
                    false, qa[2 * s2 + 1], false, bt1, (short)0, sacc1, false, false);
            }
            float* sp = &Spart[(dh * 4 + rt * 2 + kt) * 256];
#pragma unroll
            for (int v = 0; v < 8; ++v) sp[v * 32 + lane] = sacc0[v] + sacc1[v];
        }
        __syncthreads();

        // ---- phase B: online softmax + O += P @ K over this wave's D slice ----
        {
            // sweep this lane's row (ln) directly in A-layout key order:
            // sv[i] <-> key ko(i) = (i>>3)*16 + 8h + (i&7); partner lane (l^16)
            // covers the complementary 16 keys, so shfl_xor(16) completes rows.
            const float* a00 = &Spart[(0 + rt * 2 + 0) * 256];  // dh0, kt0
            const float* a10 = &Spart[(4 + rt * 2 + 0) * 256];  // dh1, kt0
            const float* a01 = &Spart[(0 + rt * 2 + 1) * 256];  // dh0, kt1
            const float* a11 = &Spart[(4 + rt * 2 + 1) * 256];  // dh1, kt1
            const float* kp0 = &Kpen[cur][8 * h];       // keys 8h..8h+7
            const float* kp1 = &Kpen[cur][16 + 8 * h];  // keys 16+8h..16+8h+7
            const int base = (ln & 7) * 32 + ((ln >> 3) & 1) * 16 + 8 * h;
            float sv[16];
            float mloc = -3.0e38f;
#pragma unroll
            for (int i = 0; i < 8; ++i) {
                const float s = a00[base + i] + a10[base + i] + kp0[i];
                sv[i] = s;
                mloc = fmaxf(mloc, s);
            }
#pragma unroll
            for (int i = 0; i < 8; ++i) {
                const float s = a01[base + i] + a11[base + i] + kp1[i];
                sv[8 + i] = s;
                mloc = fmaxf(mloc, s);
            }
            const float mblk = fmaxf(mloc, __shfl_xor(mloc, 16, 32));
            const float mnew = fmaxf(mrow, mblk);
            const float alpha = __expf(mrow - mnew);
            mrow = mnew;
            // rescale accumulator: o[t][v] belongs to row v + 8h
            float av[8];
#pragma unroll
            for (int v = 0; v < 8; ++v) av[v] = __shfl(alpha, 8 * h + v, 32);
#pragma unroll
            for (int t = 0; t < 8; ++t) {
#pragma unroll
                for (int v = 0; v < 8; ++v) o[t][v] *= av[v];
            }
            // probabilities + row sum
            float ps = 0.0f;
#pragma unroll
            for (int i = 0; i < 16; ++i) { sv[i] = __expf(sv[i] - mnew); ps += sv[i]; }
            lrow = lrow * alpha + ps + __shfl_xor(ps, 16, 32);
            // P A-tile: sv is already in A-element order -> straight pair pack
            v8u pw;
#pragma unroll
            for (int e2 = 0; e2 < 8; ++e2) pw[e2] = pack_bf16x2(sv[2 * e2], sv[2 * e2 + 1]);
            const v16bf pa = __builtin_bit_cast(v16bf, pw);
            // O_t += P @ K_blk ; B tile from KsT: lane col = dsl+16t+ln, keys 8h..8h+7, 16+8h..16+8h+7
#pragma unroll
            for (int t = 0; t < 8; ++t) {
                const unsigned short* ktp = &KsT[cur][(dsl + 16 * t + ln) * 32];
                const uint4 blo = *(const uint4*)(ktp + 8 * h);
                const uint4 bhi = *(const uint4*)(ktp + 16 + 8 * h);
                v8u bw;
                bw[0] = blo.x; bw[1] = blo.y; bw[2] = blo.z; bw[3] = blo.w;
                bw[4] = bhi.x; bw[5] = bhi.y; bw[6] = bhi.z; bw[7] = bhi.w;
                const v16bf bt = __builtin_bit_cast(v16bf, bw);
                o[t] = __builtin_amdgcn_wmma_f32_16x16x32_bf16(
                    false, pa, false, bt, (short)0, o[t], false, false);
            }
        }
    }

    // ---- normalize and store ----
    const float linv = 1.0f / lrow;
    float lv[8];
#pragma unroll
    for (int v = 0; v < 8; ++v) lv[v] = __shfl(linv, 8 * h + v, 32);
#pragma unroll
    for (int t = 0; t < 8; ++t) {
        const int d = dsl + 16 * t + ln;
#pragma unroll
        for (int v = 0; v < 8; ++v) {
            const int row = row_block + rt * 16 + v + 8 * h;
            Out[((size_t)b * Nq + row) * D + d] = o[t][v] * lv[v];
        }
    }
}

extern "C" void kernel_launch(void* const* d_in, const int* in_sizes, int n_in,
                              void* d_out, int out_size, void* d_ws, size_t ws_size,
                              hipStream_t stream) {
    (void)in_sizes; (void)n_in; (void)out_size; (void)d_ws; (void)ws_size;
    const int B = 64, N = 1024, M = 1024, D = 512;
    const float* P = (const float*)d_in[0];
    const float* H = (const float*)d_in[1];
    const unsigned char* pm = (const unsigned char*)d_in[2];
    const unsigned char* hm = (const unsigned char*)d_in[3];
    float* outP = (float*)d_out;                       // premise_aligned  [B,N,D]
    float* outH = outP + (size_t)B * N * D;            // hypothesis_aligned [B,M,D]

    dim3 blk(256);
    // premise_aligned: queries = premise, keys/values = hypothesis (softmax over M)
    coattn_bf16_wmma<<<dim3(N / 32, B), blk, 0, stream>>>(P, H, pm, hm, outP, N, M, D);
    // hypothesis_aligned: queries = hypothesis, keys/values = premise (softmax over N)
    coattn_bf16_wmma<<<dim3(M / 32, B), blk, 0, stream>>>(H, P, hm, pm, outH, M, N, D);
}
